// Mamba_9672266350944
// MI455X (gfx1250) — compile-verified, matrix operand order checked by reading, not run on hardware
//
#include <hip/hip_runtime.h>
#include <hip/hip_bf16.h>
#include <math.h>

// ---------------------------------------------------------------------------
// Mamba block (chain-tree variant) for MI455X / gfx1250, wave32 + WMMA.
//   B=2, L=1024, D_MODEL=1024, D_INNER=2048, D_STATE=16, DT_RANK=64
// GEMMs: async global->LDS (ASYNCcnt, double-buffered) -> ds_load fragments ->
// v_wmma_f32_16x16x32_bf16, 32x32 output tile per wave (4 wmma / K-step).
// Scans: fused, 16 SSM states in VGPRs per thread, shared B/C rows staged
// through LDS; no (b,d,L,n) tensors materialized.
// ---------------------------------------------------------------------------

typedef __attribute__((ext_vector_type(16))) __bf16 v16bf;
typedef __attribute__((ext_vector_type(8)))  float  v8f;

#define BATCH    2
#define SEQ      1024
#define DMODEL   1024
#define DINNER   2048
#define DSTATE   16
#define DTRANK   64
#define SSMW     96   // DT_RANK + 2*D_STATE

// ---------------- CDNA5 async copy + wait primitives -----------------------
__device__ __forceinline__ void async_copy_b128(unsigned lds_off, const void* gptr) {
  // global_load_async_to_lds_b128 vdst(LDS addr), vaddr(64b global), saddr=off
  asm volatile("global_load_async_to_lds_b128 %0, %1, off"
               :: "v"(lds_off), "v"(gptr) : "memory");
}

#if defined(__has_builtin)
#  if __has_builtin(__builtin_amdgcn_s_wait_asynccnt)
#    define WAIT_ASYNC(n) __builtin_amdgcn_s_wait_asynccnt(n)
#  else
#    define WAIT_ASYNC(n) asm volatile("s_wait_asynccnt %0" :: "i"(n) : "memory")
#  endif
#else
#  define WAIT_ASYNC(n) asm volatile("s_wait_asynccnt %0" :: "i"(n) : "memory")
#endif

__device__ __forceinline__ unsigned lds_offset_of(const void* p) {
  // Generic pointers to LDS carry the intra-LDS byte offset in their low 32b.
  return (unsigned)(unsigned long long)p;
}

// ---------------------------------------------------------------------------
// Fragment load from an LDS tile (rows padded to 36 floats -> conflict-free):
// CDNA5 16-bit A-matrix layout:
//   lane 0-15 : M=lane,    K-base = 0 ; lane 16-31: M=lane-16, K-base = 8
//   elems 0-7  -> K = base + e ; elems 8-15 -> K = base + 8 + e (+16 block)
// ---------------------------------------------------------------------------
#define TPAD 36

__device__ __forceinline__ v16bf load_frag_lds(const float (*tile)[TPAD], int row0) {
  const int lane = threadIdx.x & 31;
  const int m    = lane & 15;
  const int kb   = (lane >> 4) << 3;           // 0 or 8
  const float* p = &tile[row0 + m][kb];
  const float4 f0 = *(const float4*)(p + 0);
  const float4 f1 = *(const float4*)(p + 4);
  const float4 f2 = *(const float4*)(p + 16);
  const float4 f3 = *(const float4*)(p + 20);
  v16bf r;
  r[0]  = (__bf16)f0.x; r[1]  = (__bf16)f0.y; r[2]  = (__bf16)f0.z; r[3]  = (__bf16)f0.w;
  r[4]  = (__bf16)f1.x; r[5]  = (__bf16)f1.y; r[6]  = (__bf16)f1.z; r[7]  = (__bf16)f1.w;
  r[8]  = (__bf16)f2.x; r[9]  = (__bf16)f2.y; r[10] = (__bf16)f2.z; r[11] = (__bf16)f2.w;
  r[12] = (__bf16)f3.x; r[13] = (__bf16)f3.y; r[14] = (__bf16)f3.z; r[15] = (__bf16)f3.w;
  return r;
}

// C/D 16x16 f32 tile store per ISA layout: VGPR v -> M = v + (lane<16 ? 0:8),
// N = lane&15.  Stores are coalesced across the 16-lane groups.
__device__ __forceinline__ void store_tile_f32(float* __restrict__ base, int ld, v8f c) {
  const int lane = threadIdx.x & 31;
  const int n    = lane & 15;
  const int mo   = (lane >> 4) << 3;           // 0 or 8
#pragma unroll
  for (int v = 0; v < 8; ++v)
    base[(size_t)(v + mo) * (size_t)ld + n] = c[v];
}

__device__ __forceinline__ v8f wmma_bf16(v16bf a, v16bf b, v8f c) {
  return __builtin_amdgcn_wmma_f32_16x16x32_bf16(
      /*neg_a=*/false, a, /*neg_b=*/false, b,
      /*c_mod=*/(short)0, c, /*reuse_a=*/false, /*reuse_b=*/false);
}

// ---------------------------------------------------------------------------
// Generic Y[M][N] = X[M][K] * W[N][K]^T  (f32 memory, bf16 matrix pipe).
// Block = 256 threads = 8 waves tiled 2(M) x 4(N); block tile 64 x 128.
// K-slices of 32 are async-copied global->LDS, double-buffered:
//   issue stage j+1 | s_wait_asynccnt 6 | barrier | compute(j) | barrier
// K multiple of 32; M multiple of 64; N multiple of 32 (all call sites).
// ---------------------------------------------------------------------------
#define A_ROWS 64
#define B_ROWS 128
#define A_BUF_FLOATS (A_ROWS * TPAD)
#define B_BUF_FLOATS (B_ROWS * TPAD)

__global__ void __launch_bounds__(256)
wmma_gemm_xwt(const float* __restrict__ X, int ldx,
              const float* __restrict__ W, int ldw,
              float* __restrict__ Y, int ldy,
              int M, int N, int K) {
  __shared__ float sA[2][A_ROWS][TPAD];   // 18 KB
  __shared__ float sB[2][B_ROWS][TPAD];   // 36 KB

  const int wave = threadIdx.x >> 5;
  const int wm   = wave >> 2;             // 0..1 : 32-row strip in M
  const int wn   = wave & 3;              // 0..3 : 32-col strip in N
  const int m0b  = blockIdx.y * A_ROWS;
  const int n0b  = blockIdx.x * B_ROWS;
  const int m0   = m0b + wm * 32;
  const int n0   = n0b + wn * 32;
  const bool active = (n0 < N);           // wave-uniform (M blocks are exact)

  int nvalid = N - n0b;                   // valid W rows under this block
  if (nvalid > B_ROWS) nvalid = B_ROWS;

  const float* Xblk = X + (size_t)m0b * (size_t)ldx;
  const float* Wblk = W + (size_t)n0b * (size_t)ldw;
  const unsigned ldsA = lds_offset_of(&sA[0][0][0]);
  const unsigned ldsB = lds_offset_of(&sB[0][0][0]);
  const int t = threadIdx.x;

  // Cooperative async staging of one 32-wide K-slice (6 b128 ops / thread).
  auto stage = [&](int buf, int k0) {
#pragma unroll
    for (int i = 0; i < 2; ++i) {                     // A: 512 x b128
      const int e   = t + 256 * i;
      const int row = e >> 3;
      const int c4  = (e & 7) * 4;
      const unsigned loff = ldsA + (unsigned)(buf * A_BUF_FLOATS + row * TPAD + c4) * 4u;
      async_copy_b128(loff, Xblk + (size_t)row * (size_t)ldx + k0 + c4);
    }
#pragma unroll
    for (int i = 0; i < 4; ++i) {                     // B: 1024 x b128
      const int e   = t + 256 * i;
      const int row = e >> 3;
      const int c4  = (e & 7) * 4;
      const int rowc = (row < nvalid) ? row : (nvalid - 1);   // stay in-bounds
      const unsigned loff = ldsB + (unsigned)(buf * B_BUF_FLOATS + row * TPAD + c4) * 4u;
      async_copy_b128(loff, Wblk + (size_t)rowc * (size_t)ldw + k0 + c4);
    }
  };

  v8f c00 = {0.f,0.f,0.f,0.f,0.f,0.f,0.f,0.f};
  v8f c01 = c00, c10 = c00, c11 = c00;

  stage(0, 0);
  int buf = 0;
  for (int k0 = 0; k0 < K; k0 += 32, buf ^= 1) {
    const bool more = (k0 + 32 < K);
    if (more) {
      stage(buf ^ 1, k0 + 32);
      WAIT_ASYNC(6);        // our previous-stage copies have landed
    } else {
      WAIT_ASYNC(0);
    }
    __syncthreads();        // everyone's copies for `buf` visible

    if (active) {
      v16bf a0 = load_frag_lds(sA[buf], wm * 32 + 0);
      v16bf a1 = load_frag_lds(sA[buf], wm * 32 + 16);
      v16bf b0 = load_frag_lds(sB[buf], wn * 32 + 0);
      v16bf b1 = load_frag_lds(sB[buf], wn * 32 + 16);
      c00 = wmma_bf16(a0, b0, c00);
      c01 = wmma_bf16(a0, b1, c01);
      c10 = wmma_bf16(a1, b0, c10);
      c11 = wmma_bf16(a1, b1, c11);
    }
    __syncthreads();        // done reading `buf` before it is re-staged
  }

  if (active) {
    float* y00 = Y + (size_t)m0 * (size_t)ldy + n0;
    store_tile_f32(y00,                                 ldy, c00);
    store_tile_f32(y00 + 16,                            ldy, c01);
    store_tile_f32(y00 + (size_t)16 * (size_t)ldy,      ldy, c10);
    store_tile_f32(y00 + (size_t)16 * (size_t)ldy + 16, ldy, c11);
  }
}

// ---------------------------------------------------------------------------
// Scan kernels: one thread per (b, d) channel, 16 states in VGPRs.
//   fwd[t] = A[t]*fwd[t-1] + u[t]            -> y  = C . fwd
//   bwd[t] = A[t+1]*bwd[t+1] + u[t]          -> y += C . (bwd - u)
// with u = dt*B*h, A = exp(-exp(A_log)*dt), dt = softplus(dt_raw + bias).
// The (l, B|C) slab is shared by all 256 d-threads of a block -> staged in LDS
// (32 timesteps x 32 floats = 4KB), broadcast-read (conflict-free).
// ---------------------------------------------------------------------------
#define LCHUNK 32

__device__ __forceinline__ float softplusf(float x) {
  return (x > 20.f) ? x : log1pf(__expf(x));
}

__global__ void __launch_bounds__(256)
scan_fwd_kernel(const float* __restrict__ proj,   // (B*L, 2*DINNER): hidden | gate
                const float* __restrict__ ssm,    // (B*L, 96): dt_low | B | C
                const float* __restrict__ dtraw,  // (B*L, DINNER)
                const float* __restrict__ dt_bias,
                const float* __restrict__ A_log,
                float* __restrict__ y) {          // (B, DINNER, L)
  __shared__ float sBC[LCHUNK][2 * DSTATE];       // [li][0..15]=B, [16..31]=C

  const int id = blockIdx.x * blockDim.x + threadIdx.x;   // 0 .. B*DINNER-1
  const int b = id >> 11;
  const int d = id & (DINNER - 1);

  float a[DSTATE];
#pragma unroll
  for (int n = 0; n < DSTATE; ++n) a[n] = -__expf(A_log[d * DSTATE + n]);
  const float bias = dt_bias[d];

  float s[DSTATE];
#pragma unroll
  for (int n = 0; n < DSTATE; ++n) s[n] = 0.f;

  const float* hrow  = proj  + (size_t)b * SEQ * (2 * DINNER) + d;
  const float* dtrow = dtraw + (size_t)b * SEQ * DINNER + d;
  const float* srow  = ssm   + (size_t)b * SEQ * SSMW + DTRANK;
  float*       yrow  = y + ((size_t)b * DINNER + d) * SEQ;

  const int t = threadIdx.x;                      // cooperative loader index
  for (int lt = 0; lt < SEQ; lt += LCHUNK) {
    __syncthreads();
    // 32 steps x 32 floats = 1024 floats; 256 threads x one float4 each.
    *(float4*)&sBC[t >> 3][(t & 7) * 4] =
        *(const float4*)(srow + (size_t)(lt + (t >> 3)) * SSMW + (t & 7) * 4);
    __syncthreads();

    for (int li = 0; li < LCHUNK; ++li) {
      const int l = lt + li;
      const float h   = hrow[(size_t)l * (2 * DINNER)];
      const float dtv = softplusf(dtrow[(size_t)l * DINNER] + bias);
      float yacc = 0.f;
#pragma unroll
      for (int n = 0; n < DSTATE; ++n) {
        const float dA = __expf(a[n] * dtv);
        const float u  = dtv * sBC[li][n] * h;
        s[n] = dA * s[n] + u;
        yacc += sBC[li][DSTATE + n] * s[n];
      }
      yrow[l] = yacc;
    }
  }
}

__global__ void __launch_bounds__(256)
scan_bwd_kernel(const float* __restrict__ proj,
                const float* __restrict__ ssm,
                const float* __restrict__ dtraw,
                const float* __restrict__ dt_bias,
                const float* __restrict__ A_log,
                float* __restrict__ y) {
  __shared__ float sBC[LCHUNK][2 * DSTATE];

  const int id = blockIdx.x * blockDim.x + threadIdx.x;
  const int b = id >> 11;
  const int d = id & (DINNER - 1);

  float a[DSTATE];
#pragma unroll
  for (int n = 0; n < DSTATE; ++n) a[n] = -__expf(A_log[d * DSTATE + n]);
  const float bias = dt_bias[d];

  float s[DSTATE], ap[DSTATE];
#pragma unroll
  for (int n = 0; n < DSTATE; ++n) { s[n] = 0.f; ap[n] = 0.f; }

  const float* hrow  = proj  + (size_t)b * SEQ * (2 * DINNER) + d;
  const float* dtrow = dtraw + (size_t)b * SEQ * DINNER + d;
  const float* srow  = ssm   + (size_t)b * SEQ * SSMW + DTRANK;
  float*       yrow  = y + ((size_t)b * DINNER + d) * SEQ;

  const int t = threadIdx.x;
  for (int lt = SEQ - LCHUNK; lt >= 0; lt -= LCHUNK) {
    __syncthreads();
    *(float4*)&sBC[t >> 3][(t & 7) * 4] =
        *(const float4*)(srow + (size_t)(lt + (t >> 3)) * SSMW + (t & 7) * 4);
    __syncthreads();

    for (int li = LCHUNK - 1; li >= 0; --li) {
      const int l = lt + li;
      const float h   = hrow[(size_t)l * (2 * DINNER)];
      const float dtv = softplusf(dtrow[(size_t)l * DINNER] + bias);
      float yacc = 0.f;
#pragma unroll
      for (int n = 0; n < DSTATE; ++n) {
        const float u = dtv * sBC[li][n] * h;
        s[n] = ap[n] * s[n] + u;              // bwd[l] = A[l+1]*bwd[l+1] + u[l]
        yacc += sBC[li][DSTATE + n] * (s[n] - u);  // contribute (bwd - u) only
        ap[n] = __expf(a[n] * dtv);           // A[l] for the next (l-1) step
      }
      yrow[l] += yacc;
    }
  }
}

// ---------------------------------------------------------------------------
// Gating: scan_buf[b,l,d] = (1.3*y[b,d,l] + hidden*D[d]) * silu(gate)
// ---------------------------------------------------------------------------
__global__ void __launch_bounds__(256)
gating_kernel(const float* __restrict__ proj,
              const float* __restrict__ y,
              const float* __restrict__ Dv,
              float* __restrict__ scan_buf) {
  const int id = blockIdx.x * blockDim.x + threadIdx.x;   // 0 .. B*L*DINNER-1
  const int d  = id & (DINNER - 1);
  const int bl = id >> 11;                                // 0 .. B*L-1
  const int b  = bl >> 10;
  const int l  = bl & (SEQ - 1);

  const float h  = proj[(size_t)bl * (2 * DINNER) + d];
  const float g  = proj[(size_t)bl * (2 * DINNER) + DINNER + d];
  const float yv = y[((size_t)b * DINNER + d) * SEQ + l];
  const float silu = g / (1.f + __expf(-g));
  scan_buf[(size_t)bl * DINNER + d] = (1.3f * yv + h * Dv[d]) * silu;
}

// ---------------------------------------------------------------------------
// Launch
// ---------------------------------------------------------------------------
extern "C" void kernel_launch(void* const* d_in, const int* in_sizes, int n_in,
                              void* d_out, int out_size, void* d_ws, size_t ws_size,
                              hipStream_t stream) {
  const float* x          = (const float*)d_in[0];  // (B, L, DMODEL)
  const float* in_proj_w  = (const float*)d_in[1];  // (2*DINNER, DMODEL)
  const float* x_proj_w   = (const float*)d_in[2];  // (96, DINNER)
  const float* dt_proj_w  = (const float*)d_in[3];  // (DINNER, DTRANK)
  const float* dt_proj_b  = (const float*)d_in[4];  // (DINNER,)
  const float* A_log      = (const float*)d_in[5];  // (DINNER, DSTATE)
  const float* Dv         = (const float*)d_in[6];  // (DINNER,)
  const float* out_proj_w = (const float*)d_in[7];  // (DMODEL, DINNER)
  float*       out        = (float*)d_out;          // (B, L, DMODEL)

  char* ws = (char*)d_ws;
  const size_t PROJ_BYTES = (size_t)BATCH * SEQ * (2 * DINNER) * sizeof(float); // 32 MB
  const size_t SSM_BYTES  = (size_t)BATCH * SEQ * SSMW * sizeof(float);         // 0.75 MB
  const size_t DT_BYTES   = (size_t)BATCH * SEQ * DINNER * sizeof(float);       // 16 MB

  float* proj     = (float*)(ws);
  float* ssm      = (float*)(ws + PROJ_BYTES);
  float* dtraw    = (float*)(ws + PROJ_BYTES + SSM_BYTES);
  float* ybuf     = (float*)(ws + PROJ_BYTES + SSM_BYTES + DT_BYTES);
  float* scan_buf = dtraw;   // dt_raw is dead after the scans; reuse its 16 MB

  const int M = BATCH * SEQ;         // 2048 rows everywhere
  const dim3 blk(256);

  // 1) proj = x @ in_proj_w^T            (2048x1024)x(4096x1024)^T
  wmma_gemm_xwt<<<dim3((2 * DINNER) / 128, M / 64), blk, 0, stream>>>(
      x, DMODEL, in_proj_w, DMODEL, proj, 2 * DINNER, M, 2 * DINNER, DMODEL);

  // 2) ssm = hidden @ x_proj_w^T         (2048x2048)x(96x2048)^T   (N=96, clamped B rows)
  wmma_gemm_xwt<<<dim3(1, M / 64), blk, 0, stream>>>(
      proj, 2 * DINNER, x_proj_w, DINNER, ssm, SSMW, M, SSMW, DINNER);

  // 3) dt_raw = dt_low @ dt_proj_w^T     (2048x64)x(2048x64)^T  (dt_low = ssm[:, :64])
  wmma_gemm_xwt<<<dim3(DINNER / 128, M / 64), blk, 0, stream>>>(
      ssm, SSMW, dt_proj_w, DTRANK, dtraw, DINNER, M, DINNER, DTRANK);

  // 4/5) fused forward + backward chain-tree scans -> y (B, DINNER, L)
  scan_fwd_kernel<<<dim3((BATCH * DINNER) / 256), blk, 0, stream>>>(
      proj, ssm, dtraw, dt_proj_b, A_log, ybuf);
  scan_bwd_kernel<<<dim3((BATCH * DINNER) / 256), blk, 0, stream>>>(
      proj, ssm, dtraw, dt_proj_b, A_log, ybuf);

  // 6) gating -> scan_buf (B*L, DINNER)
  gating_kernel<<<dim3((size_t)BATCH * SEQ * DINNER / 256), blk, 0, stream>>>(
      proj, ybuf, Dv, scan_buf);

  // 7) out = scan_buf @ out_proj_w^T     (2048x2048)x(1024x2048)^T
  wmma_gemm_xwt<<<dim3(DMODEL / 128, M / 64), blk, 0, stream>>>(
      scan_buf, DINNER, out_proj_w, DINNER, out, DMODEL, M, DMODEL, DINNER);
}
